// DKernelLocalStrideSparseAttention_10814727652067
// MI455X (gfx1250) — matrix operand order repaired and checked by reading, not compiled, not run on hardware
//
#include <hip/hip_runtime.h>

typedef __attribute__((ext_vector_type(16))) _Float16 v16h;
typedef __attribute__((ext_vector_type(8)))  float    v8f;

#define S_LEN      2048
#define NH         32
#define HD         128
#define BLK        64
#define LOCAL_BLKS 16
#define VERT_STR   8
#define NEG_INF    (-1.0e30f)
#define SM_SCALE   0.08838834764831845f   // 1/sqrt(128)

// LDS row strides (in halves). Chosen so each row start and every fragment
// sub-offset is a multiple of 32 bytes -> v16h LDS loads are aligned b128 pairs.
#define QS 160   // sQ / sK : [64][160]  (320B per row)
#define VS 80    // sVt     : [128][80]  (160B per row)
#define PS 80    // sP      : per-wave [32][80]

// K-index inside a 16-bit A fragment (16x32), contiguous order matching B:
// lanes 0-15: K = hh; lanes 16-31: K = 16 + hh.
__device__ __forceinline__ int a_kidx(int hh, int g) {
  return g * 16 + hh;
}

extern "C" __global__ __launch_bounds__(64)
void DKernelLocalStrideSparseAttention_10814727652067_kernel(
    const float* __restrict__ Q, const float* __restrict__ K,
    const float* __restrict__ V, float* __restrict__ O) {
  // 20480 (sQ, reused as sVt) + 20480 (sK) + 10240 (sP) = 51200 bytes
  __shared__ __attribute__((aligned(32))) char lds[51200];
  _Float16* sQ  = (_Float16*)lds;            // [64][QS] f16 (init phase only)
  _Float16* sVt = (_Float16*)lds;            // [128][VS] f16, V transposed (loop)
  _Float16* sK  = (_Float16*)(lds + 20480);  // [64][QS] f16
  _Float16* sP  = (_Float16*)(lds + 40960);  // [2][32][PS] f16 probs

  const int qblk = blockIdx.x;
  const int h    = blockIdx.y;
  const int tid  = threadIdx.x;
  const int wave = tid >> 5;    // 2 waves, each owns 32 query rows (2 M-tiles)
  const int lane = tid & 31;
  const int g    = lane >> 4;   // lane group (0: lanes 0-15, 1: lanes 16-31)
  const int ln16 = lane & 15;

  const size_t rowstr = (size_t)NH * HD;     // 4096 floats between tokens

  // ---------------- stage Q tile (64 x 128) as f16 into LDS ----------------
  const float* Qg = Q + ((size_t)(qblk * BLK) * NH + h) * HD;
  for (int i = tid; i < 2048; i += 64) {     // 2048 float4 = 64*128 floats
    int r = i >> 5, d = (i & 31) * 4;
    const float4 v = *(const float4*)(Qg + (size_t)r * rowstr + d);
    _Float16* dst = sQ + r * QS + d;
    dst[0] = (_Float16)v.x; dst[1] = (_Float16)v.y;
    dst[2] = (_Float16)v.z; dst[3] = (_Float16)v.w;
  }
  __syncthreads();

  // --- gather Q A-fragments: 2 M-tiles x 4 D-chunks (K=32), kept in VGPRs ---
  v16h aq[2][4];
#pragma unroll
  for (int m = 0; m < 2; ++m)
#pragma unroll
    for (int c = 0; c < 4; ++c)
#pragma unroll
      for (int hh = 0; hh < 16; ++hh)
        aq[m][c][hh] =
            sQ[(wave * 32 + m * 16 + ln16) * QS + c * 32 + a_kidx(hh, g)];

  // ---------------- accumulators (flash state) ----------------
  v8f o[2][8];                     // 2 x (16 x 128) output tiles (C layout)
  float m_run[2][8], l_run[2][8];
  v8f zacc = {};
#pragma unroll
  for (int m = 0; m < 2; ++m) {
#pragma unroll
    for (int dt = 0; dt < 8; ++dt) o[m][dt] = zacc;
#pragma unroll
    for (int r = 0; r < 8; ++r) { m_run[m][r] = -3.0e38f; l_run[m][r] = 0.0f; }
  }

  // ---------------- key-block loop ----------------
  for (int kblk = 0; kblk <= qblk; ++kblk) {
    // block-uniform sparsity predicate: local window OR vertical stripe
    const bool allowed =
        ((qblk - kblk) < LOCAL_BLKS) || (((kblk + h + 1) % VERT_STR) == 0);
    if (!allowed) continue;

    __syncthreads();  // previous iteration's fragment reads are done

    // stage K (natural [key][d]) and V (transposed [d][key]) as f16
    const float* Kg = K + ((size_t)(kblk * BLK) * NH + h) * HD;
    const float* Vg = V + ((size_t)(kblk * BLK) * NH + h) * HD;
    for (int i = tid; i < 2048; i += 64) {
      int r = i >> 5, d = (i & 31) * 4;
      const float4 kv = *(const float4*)(Kg + (size_t)r * rowstr + d);
      _Float16* dst = sK + r * QS + d;
      dst[0] = (_Float16)kv.x; dst[1] = (_Float16)kv.y;
      dst[2] = (_Float16)kv.z; dst[3] = (_Float16)kv.w;
      const float4 vv = *(const float4*)(Vg + (size_t)r * rowstr + d);
      sVt[(d + 0) * VS + r] = (_Float16)vv.x;
      sVt[(d + 1) * VS + r] = (_Float16)vv.y;
      sVt[(d + 2) * VS + r] = (_Float16)vv.z;
      sVt[(d + 3) * VS + r] = (_Float16)vv.w;
    }
    __syncthreads();

    // ---- S = Q K^T : 4 key tiles x 4 D-chunks; each B fragment feeds 2 WMMAs
    v8f sc[2][4];
#pragma unroll
    for (int t = 0; t < 4; ++t) {
      v8f acc0 = {}, acc1 = {};
#pragma unroll
      for (int c = 0; c < 4; ++c) {
        // B fragment (32x16): lane -> key col, halves -> contiguous d
        const _Float16* bp = sK + (t * 16 + ln16) * QS + c * 32 + g * 16;
        v16h bk = *(const v16h*)bp;
        acc0 = __builtin_amdgcn_wmma_f32_16x16x32_f16(
            false, aq[0][c], false, bk, (short)0, acc0, false, false);
        acc1 = __builtin_amdgcn_wmma_f32_16x16x32_f16(
            false, aq[1][c], false, bk, (short)0, acc1, false, false);
      }
      sc[0][t] = acc0;
      sc[1][t] = acc1;
    }

    // -------- scale + token-causal mask on the diagonal block --------
    const bool diag = (kblk == qblk);
#pragma unroll
    for (int m = 0; m < 2; ++m)
#pragma unroll
      for (int t = 0; t < 4; ++t)
#pragma unroll
        for (int r = 0; r < 8; ++r) {
          float s = sc[m][t][r] * SM_SCALE;
          if (diag) {
            int rowM = g ? (r + 8) : r;           // q pos within 64-block
            if (t * 16 + ln16 > wave * 32 + m * 16 + rowM) s = NEG_INF;
          }
          sc[m][t][r] = s;
        }

    // -------- online softmax (rows live across 16-lane groups) --------
    _Float16* pw = sP + wave * (32 * PS);
#pragma unroll
    for (int m = 0; m < 2; ++m)
#pragma unroll
      for (int r = 0; r < 8; ++r) {
        float mx = fmaxf(fmaxf(sc[m][0][r], sc[m][1][r]),
                         fmaxf(sc[m][2][r], sc[m][3][r]));
        mx = fmaxf(mx, __shfl_xor(mx, 1, 32));
        mx = fmaxf(mx, __shfl_xor(mx, 2, 32));
        mx = fmaxf(mx, __shfl_xor(mx, 4, 32));
        mx = fmaxf(mx, __shfl_xor(mx, 8, 32));
        float mnew  = fmaxf(m_run[m][r], mx);
        float alpha = __expf(m_run[m][r] - mnew);
        m_run[m][r] = mnew;
        int rowM = g ? (r + 8) : r;
        float rs = 0.0f;
#pragma unroll
        for (int t = 0; t < 4; ++t) {
          float p = __expf(sc[m][t][r] - mnew);
          rs += p;
          pw[(m * 16 + rowM) * PS + t * 16 + ln16] = (_Float16)p;
        }
        rs += __shfl_xor(rs, 1, 32);
        rs += __shfl_xor(rs, 2, 32);
        rs += __shfl_xor(rs, 4, 32);
        rs += __shfl_xor(rs, 8, 32);
        l_run[m][r] = l_run[m][r] * alpha + rs;
#pragma unroll
        for (int dt = 0; dt < 8; ++dt) o[m][dt][r] *= alpha;
      }

    // per-wave ordering of sP stores before gathers (sP region is wave-private)
    asm volatile("s_wait_dscnt 0" ::: "memory");

    // ---- O += P V : 2 key chunks (K=32) x 8 d-tiles; B reused by 2 M-tiles -
#pragma unroll
    for (int kc = 0; kc < 2; ++kc) {
      v16h ap[2];
#pragma unroll
      for (int m = 0; m < 2; ++m)
#pragma unroll
        for (int hh = 0; hh < 16; ++hh)
          ap[m][hh] = pw[(m * 16 + ln16) * PS + kc * 32 + a_kidx(hh, g)];
#pragma unroll
      for (int dt = 0; dt < 8; ++dt) {
        // B fragment (32x16): lane -> d col, halves -> contiguous keys (V^T)
        const _Float16* bp = sVt + (dt * 16 + ln16) * VS + kc * 32 + g * 16;
        v16h bv = *(const v16h*)bp;
        o[0][dt] = __builtin_amdgcn_wmma_f32_16x16x32_f16(
            false, ap[0], false, bv, (short)0, o[0][dt], false, false);
        o[1][dt] = __builtin_amdgcn_wmma_f32_16x16x32_f16(
            false, ap[1], false, bv, (short)0, o[1][dt], false, false);
      }
    }
  }

  // ---------------- finalize: O / l, write bqhd fp32 ----------------
#pragma unroll
  for (int m = 0; m < 2; ++m) {
    float* Og = O + ((size_t)(qblk * BLK + wave * 32 + m * 16) * NH + h) * HD;
#pragma unroll
    for (int r = 0; r < 8; ++r) {
      int rowM = g ? (r + 8) : r;
      float inv = 1.0f / l_run[m][r];
#pragma unroll
      for (int dt = 0; dt < 8; ++dt)
        Og[(size_t)rowM * rowstr + dt * 16 + ln16] = o[m][dt][r] * inv;
    }
  }
}

extern "C" void kernel_launch(void* const* d_in, const int* in_sizes, int n_in,
                              void* d_out, int out_size, void* d_ws, size_t ws_size,
                              hipStream_t stream) {
  (void)in_sizes; (void)n_in; (void)out_size; (void)d_ws; (void)ws_size;
  const float* q = (const float*)d_in[0];
  const float* k = (const float*)d_in[1];
  const float* v = (const float*)d_in[2];
  float* out = (float*)d_out;
  dim3 grid(S_LEN / BLK, NH);   // 32 query blocks x 32 heads
  dim3 block(64);               // 2 wave32 per workgroup, 32 query rows each
  DKernelLocalStrideSparseAttention_10814727652067_kernel<<<grid, block, 0, stream>>>(q, k, v, out);
}
